// myxLSTM_704374637219
// MI455X (gfx1250) — matrix operand order repaired and checked by reading, not compiled, not run on hardware
//
#include <hip/hip_runtime.h>

// ============================================================================
// xLSTM forward for MI455X (gfx1250, wave32, WMMA).
// All GEMM-shaped work (proj_up/proj_down, ff_up/ff_down, attention QK^T and
// P@V, sLSTM recurrence) runs on v_wmma_f32_16x16x32_f16 (f32 accumulate).
// Scratch lives in d_ws with phase-aliased layout. Input order = dict
// insertion order of setup_inputs(), recursively flattened.
// ============================================================================

typedef __attribute__((ext_vector_type(16))) _Float16 v16h;
typedef __attribute__((ext_vector_type(8)))  float    v8f;

constexpr int kB  = 16;    // batch
constexpr int kS  = 512;   // sequence after conv stem
constexpr int kD  = 128;   // model dim
constexpr int kI  = 256;   // mLSTM inner dim
constexpr int kNH = 4;     // heads
constexpr int kT  = kB * kS; // tokens = 8192

#define DEV static __device__ __forceinline__

DEV float warp_sum(float v) {
#pragma unroll
  for (int m = 16; m > 0; m >>= 1) v += __shfl_xor(v, m, 32);
  return v;
}
DEV float sigmoidf_(float x) { return 1.f / (1.f + __expf(-x)); }
DEV float siluf_(float x)    { return x * sigmoidf_(x); }
DEV float logsigf_(float x)  { return fminf(x, 0.f) - log1pf(__expf(-fabsf(x))); }

DEV v8f wmma16(v16h a, v16h b, v8f c) {
  return __builtin_amdgcn_wmma_f32_16x16x32_f16(false, a, false, b, (short)0, c,
                                                false, false);
}

// ---- WMMA fragment loaders (ISA 7.12.2 layouts, wave32) --------------------
// A (16x32 f16): lane m=l&15, half=l>>4; vgpr r<4 -> K=8*half+2r, r>=4 -> K=16+8*half+2(r-4)
DEV v16h load_a_f32(const float* A, int lda, int row0, int k0) {
  const int lane = threadIdx.x & 31, m = lane & 15, hf = lane >> 4;
  v16h a;
#pragma unroll
  for (int r = 0; r < 8; ++r) {
    const int kk = (r < 4) ? (8 * hf + 2 * r) : (16 + 8 * hf + 2 * (r - 4));
    const float* p = A + (size_t)(row0 + m) * lda + k0 + kk;
    a[2 * r]     = (_Float16)p[0];
    a[2 * r + 1] = (_Float16)p[1];
  }
  return a;
}
DEV v16h load_a_lds_f16(const _Float16* ct) {  // ct: 16x32 row-major
  const int lane = threadIdx.x & 31, m = lane & 15, hf = lane >> 4;
  v16h a;
#pragma unroll
  for (int r = 0; r < 8; ++r) {
    const int kk = (r < 4) ? (8 * hf + 2 * r) : (16 + 8 * hf + 2 * (r - 4));
    a[2 * r]     = ct[m * 32 + kk];
    a[2 * r + 1] = ct[m * 32 + kk + 1];
  }
  return a;
}
DEV v16h load_a_lds_f32(const float* ct) {  // ct: 16x32 row-major f32 (LDS)
  const int lane = threadIdx.x & 31, m = lane & 15, hf = lane >> 4;
  v16h a;
#pragma unroll
  for (int r = 0; r < 8; ++r) {
    const int kk = (r < 4) ? (8 * hf + 2 * r) : (16 + 8 * hf + 2 * (r - 4));
    a[2 * r]     = (_Float16)ct[m * 32 + kk];
    a[2 * r + 1] = (_Float16)ct[m * 32 + kk + 1];
  }
  return a;
}
// B (32x16 f16): lane n=l&15, half=l>>4; vgpr r<4 -> K=8*half+2r, r>=4 -> K=16+8*half+2(r-4)
DEV v16h load_b_f16(const _Float16* Bm, int ldb, int k0, int col0) {
  const int lane = threadIdx.x & 31, n = lane & 15, hf = lane >> 4;
  v16h b;
#pragma unroll
  for (int r = 0; r < 8; ++r) {
    const int kk = (r < 4) ? (8 * hf + 2 * r) : (16 + 8 * hf + 2 * (r - 4));
    b[2 * r]     = Bm[(size_t)(k0 + kk) * ldb + col0 + n];
    b[2 * r + 1] = Bm[(size_t)(k0 + kk + 1) * ldb + col0 + n];
  }
  return b;
}
DEV v16h load_b_f32(const float* Bm, int ldb, int k0, int col0) {
  const int lane = threadIdx.x & 31, n = lane & 15, hf = lane >> 4;
  v16h b;
#pragma unroll
  for (int r = 0; r < 8; ++r) {
    const int kk = (r < 4) ? (8 * hf + 2 * r) : (16 + 8 * hf + 2 * (r - 4));
    b[2 * r]     = (_Float16)Bm[(size_t)(k0 + kk) * ldb + col0 + n];
    b[2 * r + 1] = (_Float16)Bm[(size_t)(k0 + kk + 1) * ldb + col0 + n];
  }
  return b;
}
// B[k][n] = Sm[col0+n][k0+k]  (transposed source, contiguous along k)
DEV v16h load_bT_f32(const float* Sm, int lds, int k0, int col0) {
  const int lane = threadIdx.x & 31, n = lane & 15, hf = lane >> 4;
  v16h b;
#pragma unroll
  for (int r = 0; r < 8; ++r) {
    const int kk = (r < 4) ? (8 * hf + 2 * r) : (16 + 8 * hf + 2 * (r - 4));
    const float* p = Sm + (size_t)(col0 + n) * lds + k0 + kk;
    b[2 * r]     = (_Float16)p[0];
    b[2 * r + 1] = (_Float16)p[1];
  }
  return b;
}

// ============================== generic kernels =============================

__global__ void cvt16(const float* __restrict__ s, _Float16* __restrict__ d, int n) {
  int i = blockIdx.x * blockDim.x + threadIdx.x;
  if (i < n) d[i] = (_Float16)s[i];
}

__global__ void addk(const float* __restrict__ a, const float* __restrict__ b,
                     float* __restrict__ o, int n) {
  int i = blockIdx.x * blockDim.x + threadIdx.x;
  if (i < n) o[i] = a[i] + b[i];
}

__global__ void revk(const float* __restrict__ src, float* __restrict__ dst,
                     int B_, int S, int D) {
  int idx = blockIdx.x * blockDim.x + threadIdx.x;
  if (idx >= B_ * S * D) return;
  int d = idx % D, s = (idx / D) % S, b = idx / (D * S);
  dst[((size_t)b * S + s) * D + d] = src[((size_t)b * S + (S - 1 - s)) * D + d];
}

__global__ void transp(const float* __restrict__ in, float* __restrict__ out,
                       int B_, int C, int L) {  // (B,C,L) -> (B,L,C)
  int idx = blockIdx.x * blockDim.x + threadIdx.x;
  if (idx >= B_ * C * L) return;
  int l = idx % L, c = (idx / L) % C, b = idx / (L * C);
  out[((size_t)b * L + l) * C + c] = in[((size_t)b * C + c) * L + l];
}

__global__ void splitk(const float* __restrict__ proj, float* __restrict__ a,
                       float* __restrict__ b, int T, int C) {
  int idx = blockIdx.x * blockDim.x + threadIdx.x;
  if (idx >= T * C) return;
  int t = idx / C, c = idx - t * C;
  a[idx] = proj[(size_t)t * (2 * C) + c];
  b[idx] = proj[(size_t)t * (2 * C) + C + c];
}

// one wave per token; out = LN(x (+add)) * w
__global__ void ln_k(const float* __restrict__ x, const float* __restrict__ add,
                     const float* __restrict__ w, float* __restrict__ out,
                     int T, int D) {
  int t = blockIdx.x * (blockDim.x >> 5) + (threadIdx.x >> 5);
  if (t >= T) return;
  int lane = threadIdx.x & 31;
  float vals[4];
  float s = 0.f;
  for (int i = 0; i < D / 32; ++i) {
    float v = x[(size_t)t * D + lane + 32 * i];
    if (add) v += add[(size_t)t * D + lane + 32 * i];
    vals[i] = v; s += v;
  }
  s = warp_sum(s);
  float mu = s / D, q = 0.f;
  for (int i = 0; i < D / 32; ++i) { float d0 = vals[i] - mu; q += d0 * d0; }
  q = warp_sum(q);
  float rstd = rsqrtf(q / D + 1e-5f);
  for (int i = 0; i < D / 32; ++i)
    out[(size_t)t * D + lane + 32 * i] = (vals[i] - mu) * rstd * w[lane + 32 * i];
}

// C[MxN] = A[MxK](f32) @ W[KxN](f16), one 16x16 tile per wave
__global__ void gemm16(const float* __restrict__ A, const _Float16* __restrict__ W,
                       float* __restrict__ C, int M, int K, int N) {
  const int wid = blockIdx.x * (blockDim.x >> 5) + (threadIdx.x >> 5);
  const int ntiles = N >> 4;
  const int mt = wid / ntiles, nt = wid - mt * ntiles;
  if (mt >= (M >> 4)) return;
  v8f acc = {};
  for (int k0 = 0; k0 < K; k0 += 32) {
    if (k0 + 32 < K)
      __builtin_prefetch(A + (size_t)(mt * 16 + (threadIdx.x & 15)) * K + k0 + 32, 0, 1);
    v16h a = load_a_f32(A, K, mt * 16, k0);
    v16h b = load_b_f16(W, N, k0, nt * 16);
    acc = wmma16(a, b, acc);
  }
  const int lane = threadIdx.x & 31, n = lane & 15, hf = lane >> 4;
#pragma unroll
  for (int v = 0; v < 8; ++v)
    C[(size_t)(mt * 16 + v + 8 * hf) * N + nt * 16 + n] = acc[v];
}

// =============================== conv stem ==================================
__global__ void conv_bn_pool(const float* __restrict__ in, int Cin, int Lin, int mode,
                             const float* __restrict__ w, const float* __restrict__ bc,
                             const float* __restrict__ bg, const float* __restrict__ bb,
                             const float* __restrict__ bm, const float* __restrict__ bv,
                             float* __restrict__ out, int Cout, int B_) {
  int Lout = Lin >> 1;
  int idx = blockIdx.x * blockDim.x + threadIdx.x;
  if (idx >= B_ * Cout * Lout) return;
  int lo = idx % Lout, co = (idx / Lout) % Cout, b = idx / (Lout * Cout);
  float scale = rsqrtf(bv[co] + 1e-5f) * bg[co];
  float mx = -1e30f;
  for (int p = 0; p < 2; ++p) {
    int l = 2 * lo + p;
    float acc = bc[co];
    for (int ci = 0; ci < Cin; ++ci) {
      const float* wr = w + (co * Cin + ci) * 3;
      for (int kk = 0; kk < 3; ++kk) {
        int li = l + kk - 1;
        if (li < 0 || li >= Lin) continue;
        float xv = (mode == 0) ? in[((size_t)b * Lin + li) * Cin + ci]
                               : in[((size_t)b * Cin + ci) * Lin + li];
        acc += wr[kk] * xv;
      }
    }
    float h = (acc - bm[co]) * scale + bb[co];
    h = (h >= 0.f) ? h : 0.01f * h;
    mx = fmaxf(mx, h);
  }
  out[((size_t)b * Cout + co) * Lout + lo] = mx;
}

// =============================== mLSTM ======================================
__global__ void dwconv_silu(const float* __restrict__ x, const float* __restrict__ w,
                            const float* __restrict__ b, float* __restrict__ out,
                            int B_, int S, int C) {
  int idx = blockIdx.x * blockDim.x + threadIdx.x;
  if (idx >= B_ * S * C) return;
  int c = idx % C, s = (idx / C) % S, bb = idx / (C * S);
  const float* xb = x + (size_t)bb * S * C;
  float acc = b[c] + w[c * 3 + 2] * xb[(size_t)s * C + c];
  if (s >= 1) acc += w[c * 3 + 1] * xb[(size_t)(s - 1) * C + c];
  if (s >= 2) acc += w[c * 3 + 0] * xb[(size_t)(s - 2) * C + c];
  out[idx] = siluf_(acc);
}

__global__ void headwise4(const float* __restrict__ x, const float* __restrict__ w,
                          float* __restrict__ out, int T) {  // (NB=64,4,4) block-diag
  int idx = blockIdx.x * blockDim.x + threadIdx.x;
  if (idx >= T * 64) return;
  int nb = idx & 63;
  size_t t = idx >> 6;
  const float* xi = x + t * kI + nb * 4;
  const float* wn = w + nb * 16;
#pragma unroll
  for (int o = 0; o < 4; ++o) {
    float a = 0.f;
#pragma unroll
    for (int i = 0; i < 4; ++i) a += xi[i] * wn[o * 4 + i];
    out[t * kI + nb * 4 + o] = a;
  }
}

// ig/fg gates: one wave per token, ifin = [q|k|v] (768)
__global__ void mgates(const float* __restrict__ q, const float* __restrict__ k,
                       const float* __restrict__ v, const float* __restrict__ igw,
                       const float* __restrict__ igb, const float* __restrict__ fgw,
                       const float* __restrict__ fgb, float* __restrict__ ig,
                       float* __restrict__ fg, int B_, int S) {
  int t = blockIdx.x * (blockDim.x >> 5) + (threadIdx.x >> 5);
  if (t >= B_ * S) return;
  int lane = threadIdx.x & 31;
  float aI[4] = {0, 0, 0, 0}, aF[4] = {0, 0, 0, 0};
  for (int i = lane; i < 768; i += 32) {
    float xv = (i < 256) ? q[(size_t)t * 256 + i]
             : (i < 512) ? k[(size_t)t * 256 + i - 256]
                         : v[(size_t)t * 256 + i - 512];
#pragma unroll
    for (int g = 0; g < 4; ++g) {
      aI[g] += xv * igw[i * 4 + g];
      aF[g] += xv * fgw[i * 4 + g];
    }
  }
#pragma unroll
  for (int g = 0; g < 4; ++g) { aI[g] = warp_sum(aI[g]); aF[g] = warp_sum(aF[g]); }
  if (lane == 0) {
    int b = t / S, s = t - b * S;
#pragma unroll
    for (int g = 0; g < 4; ++g) {
      ig[((size_t)(b * 4 + g)) * S + s] = aI[g] + igb[g];
      fg[((size_t)(b * 4 + g)) * S + s] = aF[g] + fgb[g];
    }
  }
}

// per (b,head) prefix work: g[t]=ig[t]-cumsum(logsig(fg)), M[t]=running max g,
// elim[t]=exp(-(cs+M)) = exp(-maxD)
__global__ void mscan(const float* __restrict__ fg, const float* __restrict__ ig,
                      float* __restrict__ g, float* __restrict__ Mx,
                      float* __restrict__ elim, int BH, int S) {
  int i = blockIdx.x * blockDim.x + threadIdx.x;
  if (i >= BH) return;
  const float* f = fg + (size_t)i * S;
  const float* ii = ig + (size_t)i * S;
  float c = 0.f, M = -1e30f;
  for (int s = 0; s < S; ++s) {
    c += logsigf_(f[s]);
    float gg = ii[s] - c;
    M = fmaxf(M, gg);
    g[(size_t)i * S + s] = gg;
    Mx[(size_t)i * S + s] = M;
    elim[(size_t)i * S + s] = __expf(-(c + M));
  }
}

// masked-decay attention; one wave per (b, head, 16-row query tile)
__global__ void mattn(const float* __restrict__ q, const float* __restrict__ k,
                      const float* __restrict__ v, const float* __restrict__ g,
                      const float* __restrict__ Mx, const float* __restrict__ elim,
                      float* __restrict__ hh, int S) {
  __shared__ _Float16 cst[4 * 16 * 32];
  const int wave = threadIdx.x >> 5, lane = threadIdx.x & 31;
  const int wid = blockIdx.x * (blockDim.x >> 5) + wave;
  const int nsb = S >> 4;
  const int sb = wid % nsb, bh = wid / nsb;
  const int b = bh >> 2, hd = bh & 3;
  const float* qb_ = q + ((size_t)b * S) * kI + hd * 64;
  const float* kb_ = k + ((size_t)b * S) * kI + hd * 64;
  const float* vb_ = v + ((size_t)b * S) * kI + hd * 64;
  const float* gb  = g + (size_t)bh * S;
  const float* Mb  = Mx + (size_t)bh * S;
  const float* eb  = elim + (size_t)bh * S;
  _Float16* ct = cst + wave * 512;
  const int n = lane & 15, hf = lane >> 4;

  v16h qf0 = load_a_f32(qb_, kI, sb * 16, 0);
  v16h qf1 = load_a_f32(qb_, kI, sb * 16, 32);

  float Mrow[8], rowacc[8];
#pragma unroll
  for (int vv = 0; vv < 8; ++vv) {
    Mrow[vv] = Mb[sb * 16 + vv + 8 * hf];
    rowacc[vv] = 0.f;
  }
  v8f zz = {};
  v8f hacc[4] = {zz, zz, zz, zz};

  for (int tb0 = 0; tb0 <= sb; tb0 += 2) {
#pragma unroll
    for (int j = 0; j < 2; ++j) {
      const int tb = tb0 + j;
      v8f c = {};
      c = wmma16(qf0, load_bT_f32(kb_, kI, 0, tb * 16), c);
      c = wmma16(qf1, load_bT_f32(kb_, kI, 32, tb * 16), c);
#pragma unroll
      for (int vv = 0; vv < 8; ++vv) {
        const int t = tb * 16 + n;
        const int s = sb * 16 + vv + 8 * hf;
        float fac = (t <= s) ? __expf(gb[t] - Mrow[vv]) * 0.125f : 0.f;  // 1/sqrt(64)
        float val = c[vv] * fac;
        rowacc[vv] += val;
        ct[(vv + 8 * hf) * 32 + j * 16 + n] = (_Float16)val;
      }
    }
    // wave-local LDS exchange (LDS in-order per wave; make writes visible)
    asm volatile("s_wait_dscnt 0" ::: "memory");
    v16h af = load_a_lds_f16(ct);
#pragma unroll
    for (int dt = 0; dt < 4; ++dt)
      hacc[dt] = wmma16(af, load_b_f32(vb_, kI, tb0 * 16, dt * 16), hacc[dt]);
  }
#pragma unroll
  for (int vv = 0; vv < 8; ++vv) {
    float r = rowacc[vv];
#pragma unroll
    for (int msk = 1; msk < 16; msk <<= 1) r += __shfl_xor(r, msk, 32);
    const int s = sb * 16 + vv + 8 * hf;
    const float inv = 1.f / (fmaxf(fabsf(r), eb[s]) + 1e-6f);
    float* hr = hh + ((size_t)b * S + s) * kI + hd * 64;
#pragma unroll
    for (int dt = 0; dt < 4; ++dt) hr[dt * 16 + n] = hacc[dt][vv] * inv;
  }
}

// group-norm(64)/skip/silu-gate; one wave per (token, head)
__global__ void mpost(const float* __restrict__ hh, const float* __restrict__ xc,
                      const float* __restrict__ z, const float* __restrict__ nw,
                      const float* __restrict__ skip, float* __restrict__ y, int T) {
  int g = blockIdx.x * (blockDim.x >> 5) + (threadIdx.x >> 5);
  if (g >= T * 4) return;
  int head = g & 3;
  size_t base = (size_t)(g >> 2) * kI + head * 64;
  int lane = threadIdx.x & 31;
  float v0 = hh[base + lane], v1 = hh[base + 32 + lane];
  float mu = warp_sum(v0 + v1) * (1.f / 64.f);
  float d0 = v0 - mu, d1 = v1 - mu;
  float rstd = rsqrtf(warp_sum(d0 * d0 + d1 * d1) * (1.f / 64.f) + 1e-5f);
  int c0 = head * 64 + lane, c1 = c0 + 32;
  float h0 = d0 * rstd * nw[c0] + skip[c0] * xc[base + lane];
  float h1 = d1 * rstd * nw[c1] + skip[c1] * xc[base + 32 + lane];
  y[base + lane]      = h0 * siluf_(z[base + lane]);
  y[base + 32 + lane] = h1 * siluf_(z[base + 32 + lane]);
}

// =============================== sLSTM ======================================
__global__ void sheads(const float* __restrict__ xc, const float* __restrict__ h1,
                       const float* __restrict__ iw, const float* __restrict__ fw,
                       const float* __restrict__ zw, const float* __restrict__ ow,
                       float* __restrict__ iin, float* __restrict__ fin,
                       float* __restrict__ zin, float* __restrict__ oin, int T) {
  int idx = blockIdx.x * blockDim.x + threadIdx.x;
  if (idx >= T * kD) return;
  const int d = idx & 31, hd = (idx >> 5) & 3;
  const size_t t = idx >> 7;
  const float* xv = xc + t * kD + hd * 32;
  const float* hv = h1 + t * kD + hd * 32;
  const float* wi = iw + (size_t)(hd * 32 + d) * 32;
  const float* wf = fw + (size_t)(hd * 32 + d) * 32;
  const float* wz = zw + (size_t)(hd * 32 + d) * 32;
  const float* wo = ow + (size_t)(hd * 32 + d) * 32;
  float ai = 0, af = 0, az = 0, ao = 0;
#pragma unroll 8
  for (int i = 0; i < 32; ++i) {
    ai += xv[i] * wi[i]; af += xv[i] * wf[i];
    az += hv[i] * wz[i]; ao += hv[i] * wo[i];
  }
  iin[idx] = ai; fin[idx] = af; zin[idx] = az; oin[idx] = ao;
}

// sequential 512-step scan; one block per head; 8 waves each compute one
// 16(batch)x16 tile of rec = h_prev(16x32) @ R[h](32x128) via WMMA each step.
__global__ void sscan(const float* __restrict__ iin, const float* __restrict__ fin,
                      const float* __restrict__ zin, const float* __restrict__ oin,
                      const _Float16* __restrict__ R16, const float* __restrict__ bias,
                      float* __restrict__ ys, int B_, int S) {
  __shared__ float hsh[16 * 32], csh[16 * 32], nsh[16 * 32], msh[16 * 32];
  __shared__ float rec[16 * 128];
  const int head = blockIdx.x, tid = threadIdx.x;
  const int wave = tid >> 5;
  for (int i = tid; i < 512; i += 256) {
    hsh[i] = 0.f; csh[i] = 0.f; nsh[i] = 0.f; msh[i] = -1e30f;
  }
  __syncthreads();
  const _Float16* Rh = R16 + (size_t)head * 32 * 128;
  const v16h bfr = load_b_f16(Rh, 128, 0, wave * 16);  // constant across steps
  const int lane = tid & 31, n = lane & 15, hf = lane >> 4;
  for (int s = 0; s < S; ++s) {
    v16h a = load_a_lds_f32(hsh);
    v8f acc = {};
    acc = wmma16(a, bfr, acc);
#pragma unroll
    for (int v = 0; v < 8; ++v)
      rec[(v + 8 * hf) * 128 + wave * 16 + n] = acc[v];
    __syncthreads();
    for (int i = tid; i < 512; i += 256) {
      int b = i >> 5, d = i & 31;
      size_t gofs = ((size_t)b * S + s) * kD + head * 32 + d;
      float ig  = iin[gofs] + rec[b * 128 +  0 + d] + bias[(head * 4 + 0) * 32 + d];
      float fgv = fin[gofs] + rec[b * 128 + 32 + d] + bias[(head * 4 + 1) * 32 + d];
      float zg  = zin[gofs] + rec[b * 128 + 64 + d] + bias[(head * 4 + 2) * 32 + d];
      float og  = oin[gofs] + rec[b * 128 + 96 + d] + bias[(head * 4 + 3) * 32 + d];
      float lfm = msh[i] + logsigf_(fgv);
      float mn  = fmaxf(ig, lfm);
      float ie  = __expf(ig - mn), fe = __expf(lfm - mn);
      float cn  = fe * csh[i] + ie * tanhf(zg);
      float nn  = fe * nsh[i] + ie;
      float hn  = sigmoidf_(og) * cn / nn;
      csh[i] = cn; nsh[i] = nn; msh[i] = mn; hsh[i] = hn;
      ys[gofs] = hn;
    }
    __syncthreads();
  }
}

// group-norm(32)*gn_w + residual; one wave per (token, head)
__global__ void spost_gn(const float* __restrict__ ys, const float* __restrict__ gnw,
                         const float* __restrict__ x, float* __restrict__ x2, int T) {
  int g = blockIdx.x * (blockDim.x >> 5) + (threadIdx.x >> 5);
  if (g >= T * 4) return;
  int head = g & 3;
  size_t base = (size_t)(g >> 2) * kD + head * 32;
  int lane = threadIdx.x & 31;
  float v = ys[base + lane];
  float mu = warp_sum(v) * (1.f / 32.f);
  float d = v - mu;
  float rstd = rsqrtf(warp_sum(d * d) * (1.f / 32.f) + 1e-5f);
  x2[base + lane] = x[base + lane] + d * rstd * gnw[head * 32 + lane];
}

__global__ void geluglu(const float* __restrict__ u, float* __restrict__ v, int T) {
  int idx = blockIdx.x * blockDim.x + threadIdx.x;
  if (idx >= T * 192) return;
  int t = idx / 192, j = idx - t * 192;
  float gate = u[(size_t)t * 384 + j], up = u[(size_t)t * 384 + 192 + j];
  float ge = 0.5f * gate * (1.f + erff(gate * 0.70710678f));
  v[idx] = ge * up;
}

// =============================== head =======================================
__global__ void headk(const float* __restrict__ e, const float* __restrict__ w1,
                      const float* __restrict__ b1, const float* __restrict__ w2,
                      const float* __restrict__ b2, float* __restrict__ out, int S) {
  __shared__ float hbuf[16 * 64];
  int tid = threadIdx.x;
  for (int i = tid; i < 16 * 64; i += blockDim.x) {
    int b = i >> 6, j = i & 63;
    const float* er = e + ((size_t)b * S + (S - 1)) * kD;
    float a = b1[j];
    for (int kk = 0; kk < 128; ++kk) a += er[kk] * w1[kk * 64 + j];
    hbuf[i] = (a >= 0.f) ? a : 0.01f * a;
  }
  __syncthreads();
  for (int i = tid; i < 16 * 5; i += blockDim.x) {
    int b = i / 5, j = i - b * 5;
    float a = b2[j];
    for (int kk = 0; kk < 64; ++kk) a += hbuf[b * 64 + kk] * w2[kk * 5 + j];
    out[i] = a;
  }
}

// =============================== host side ==================================
struct MScr {
  float *xm, *zb, *xcm, *qb, *kb, *vb, *hhb, *yb, *igb, *fgb, *gbb, *Mxb, *elb, *pdb;
  _Float16 *pu16, *pdw16;
};
struct SScr {
  float *sh1, *sxc, *iin, *fin, *zin, *oin, *ysb, *x2b, *tb, *ub, *gub, *t2b;
  _Float16 *fu16, *fd16, *R16;
};

static void run_mlstm(hipStream_t st, const float* x, float* out,
                      const float* const* p, const MScr& w) {
  const int T = kT;
  cvt16<<<(128 * 512 + 255) / 256, 256, 0, st>>>(p[1], w.pu16, 128 * 512);
  cvt16<<<(256 * 128 + 255) / 256, 256, 0, st>>>(p[13], w.pdw16, 256 * 128);
  ln_k<<<T / 4, 128, 0, st>>>(x, nullptr, p[0], w.pdb, T, kD);
  gemm16<<<((T / 16) * 32) / 4, 128, 0, st>>>(w.pdb, w.pu16, w.hhb, T, 128, 512);
  splitk<<<(T * 256 + 255) / 256, 256, 0, st>>>(w.hhb, w.xm, w.zb, T, 256);
  dwconv_silu<<<(T * 256 + 255) / 256, 256, 0, st>>>(w.xm, p[2], p[3], w.xcm, kB, kS, 256);
  headwise4<<<(T * 64 + 255) / 256, 256, 0, st>>>(w.xcm, p[4], w.qb, T);
  headwise4<<<(T * 64 + 255) / 256, 256, 0, st>>>(w.xcm, p[5], w.kb, T);
  headwise4<<<(T * 64 + 255) / 256, 256, 0, st>>>(w.xm, p[6], w.vb, T);
  mgates<<<T / 4, 128, 0, st>>>(w.qb, w.kb, w.vb, p[7], p[8], p[9], p[10],
                                w.igb, w.fgb, kB, kS);
  mscan<<<1, 64, 0, st>>>(w.fgb, w.igb, w.gbb, w.Mxb, w.elb, kB * kNH, kS);
  mattn<<<(kB * kNH * (kS / 16)) / 4, 128, 0, st>>>(w.qb, w.kb, w.vb, w.gbb,
                                                    w.Mxb, w.elb, w.hhb, kS);
  mpost<<<(T * kNH) / 4, 128, 0, st>>>(w.hhb, w.xcm, w.zb, p[12], p[11], w.yb, T);
  gemm16<<<((T / 16) * 8) / 4, 128, 0, st>>>(w.yb, w.pdw16, w.pdb, T, 256, 128);
  ln_k<<<T / 4, 128, 0, st>>>(x, w.pdb, p[14], out, T, kD);
}

static void run_slstm(hipStream_t st, const float* x, float* out,
                      const float* const* p, const SScr& w) {
  const int T = kT;
  cvt16<<<(128 * 384 + 255) / 256, 256, 0, st>>>(p[11], w.fu16, 128 * 384);
  cvt16<<<(192 * 128 + 255) / 256, 256, 0, st>>>(p[12], w.fd16, 192 * 128);
  cvt16<<<(4 * 32 * 128 + 255) / 256, 256, 0, st>>>(p[7], w.R16, 4 * 32 * 128);
  ln_k<<<T / 4, 128, 0, st>>>(x, nullptr, p[0], w.sh1, T, kD);
  dwconv_silu<<<(T * 128 + 255) / 256, 256, 0, st>>>(w.sh1, p[1], p[2], w.sxc, kB, kS, 128);
  sheads<<<(T * 128 + 255) / 256, 256, 0, st>>>(w.sxc, w.sh1, p[3], p[4], p[5], p[6],
                                                w.iin, w.fin, w.zin, w.oin, T);
  sscan<<<4, 256, 0, st>>>(w.iin, w.fin, w.zin, w.oin, w.R16, p[8], w.ysb, kB, kS);
  spost_gn<<<(T * 4) / 4, 128, 0, st>>>(w.ysb, p[9], x, w.x2b, T);
  ln_k<<<T / 4, 128, 0, st>>>(w.x2b, nullptr, p[10], w.tb, T, kD);
  gemm16<<<((T / 16) * 24) / 4, 128, 0, st>>>(w.tb, w.fu16, w.ub, T, 128, 384);
  geluglu<<<(T * 192 + 255) / 256, 256, 0, st>>>(w.ub, w.gub, T);
  gemm16<<<((T / 16) * 8) / 4, 128, 0, st>>>(w.gub, w.fd16, w.t2b, T, 192, 128);
  ln_k<<<T / 4, 128, 0, st>>>(w.x2b, w.t2b, p[13], out, T, kD);
}

extern "C" void kernel_launch(void* const* d_in, const int* in_sizes, int n_in,
                              void* d_out, int out_size, void* d_ws, size_t ws_size,
                              hipStream_t stream) {
  (void)in_sizes; (void)n_in; (void)out_size; (void)ws_size;
  // --- input unpack: dict insertion order: x, conv(18), mlstm_a(15),
  //     mlstm_b(15), slstm_a(14), slstm_b(14), fc1_w, fc1_b, fc2_w, fc2_b ---
  int pi = 0;
  const float* xin = (const float*)d_in[pi++];
  const float* cv[18]; for (int i = 0; i < 18; ++i) cv[i] = (const float*)d_in[pi++];
  const float* ma[15]; for (int i = 0; i < 15; ++i) ma[i] = (const float*)d_in[pi++];
  const float* mbp[15]; for (int i = 0; i < 15; ++i) mbp[i] = (const float*)d_in[pi++];
  const float* sa[14]; for (int i = 0; i < 14; ++i) sa[i] = (const float*)d_in[pi++];
  const float* sbp[14]; for (int i = 0; i < 14; ++i) sbp[i] = (const float*)d_in[pi++];
  const float* fc1w = (const float*)d_in[pi++];
  const float* fc1b = (const float*)d_in[pi++];
  const float* fc2w = (const float*)d_in[pi++];
  const float* fc2b = (const float*)d_in[pi++];

  float* W = (float*)d_ws;
  const size_t BSD = (size_t)kT * kD;   // 1,048,576 floats
  const size_t BSI = (size_t)kT * kI;   // 2,097,152 floats
  const size_t BHS = (size_t)kB * kNH * kS;
  float* h    = W;
  float* macc = W + BSD;
  float* P    = W + 2 * BSD;            // phase-aliased region

  // ---- phase 1: conv stem ----
  {
    float* c0 = P;
    float* c1 = P + 1048576;
    float* c2 = c1 + 1048576;
    conv_bn_pool<<<(kB * 32 * 2048 + 255) / 256, 256, 0, stream>>>(
        xin, 2, 4096, 0, cv[0], cv[1], cv[2], cv[3], cv[4], cv[5], c0, 32, kB);
    conv_bn_pool<<<(kB * 64 * 1024 + 255) / 256, 256, 0, stream>>>(
        c0, 32, 2048, 1, cv[6], cv[7], cv[8], cv[9], cv[10], cv[11], c1, 64, kB);
    conv_bn_pool<<<(kB * 128 * 512 + 255) / 256, 256, 0, stream>>>(
        c1, 64, 1024, 1, cv[12], cv[13], cv[14], cv[15], cv[16], cv[17], c2, 128, kB);
    transp<<<(kB * 128 * 512 + 255) / 256, 256, 0, stream>>>(c2, h, kB, 128, 512);
  }

  // ---- phase 2: bidirectional mLSTM ----
  {
    float* hrev  = P;
    float* bufA  = P + BSD;
    float* bufB  = bufA + BSD;
    float* bufBr = bufB + BSD;
    MScr m;
    m.xm  = bufBr + BSD;  m.zb  = m.xm + BSI;  m.xcm = m.zb + BSI;
    m.qb  = m.xcm + BSI;  m.kb  = m.qb + BSI;  m.vb  = m.kb + BSI;
    m.hhb = m.vb + BSI;   m.yb  = m.hhb + BSI;        // hhb..yb doubles as proj buf
    m.igb = m.yb + BSI;   m.fgb = m.igb + BHS; m.gbb = m.fgb + BHS;
    m.Mxb = m.gbb + BHS;  m.elb = m.Mxb + BHS; m.pdb = m.elb + BHS;
    m.pu16  = (_Float16*)(m.pdb + BSD);
    m.pdw16 = m.pu16 + 128 * 512;

    run_mlstm(stream, h, bufA, ma, m);
    revk<<<(int)((BSD + 255) / 256), 256, 0, stream>>>(h, hrev, kB, kS, kD);
    run_mlstm(stream, hrev, bufBr, mbp, m);
    revk<<<(int)((BSD + 255) / 256), 256, 0, stream>>>(bufBr, bufB, kB, kS, kD);
    addk<<<(int)((BSD + 255) / 256), 256, 0, stream>>>(bufA, bufB, macc, (int)BSD);
  }

  // ---- phase 3: bidirectional sLSTM + head ----
  {
    float* srev   = P;
    float* soutA  = P + BSD;
    float* soutB  = soutA + BSD;
    float* soutBr = soutB + BSD;
    SScr s;
    s.sh1 = soutBr + BSD; s.sxc = s.sh1 + BSD;
    s.iin = s.sxc + BSD;  s.fin = s.iin + BSD;
    s.zin = s.fin + BSD;  s.oin = s.zin + BSD;
    s.ysb = s.oin + BSD;  s.x2b = s.ysb + BSD;  s.tb = s.x2b + BSD;
    s.ub  = s.tb + BSD;                   // T x 384
    s.gub = s.ub + (size_t)kT * 384;      // T x 192
    s.t2b = s.gub + (size_t)kT * 192;
    s.fu16 = (_Float16*)(s.t2b + BSD);
    s.fd16 = s.fu16 + 128 * 384;
    s.R16  = s.fd16 + 192 * 128;

    run_slstm(stream, macc, soutA, sa, s);
    revk<<<(int)((BSD + 255) / 256), 256, 0, stream>>>(macc, srev, kB, kS, kD);
    run_slstm(stream, srev, soutBr, sbp, s);
    revk<<<(int)((BSD + 255) / 256), 256, 0, stream>>>(soutBr, soutB, kB, kS, kD);
    addk<<<(int)((BSD + 255) / 256), 256, 0, stream>>>(soutA, soutB, s.sh1, (int)BSD);
    headk<<<1, 256, 0, stream>>>(s.sh1, fc1w, fc1b, fc2w, fc2b, (float*)d_out, kS);
  }
}
